// Attn_76888504533326
// MI455X (gfx1250) — compile-verified, matrix operand order checked by reading, not run on hardware
//
#include <hip/hip_runtime.h>

typedef __attribute__((ext_vector_type(16))) _Float16 v16h;
typedef __attribute__((ext_vector_type(8)))  _Float16 v8h;
typedef __attribute__((ext_vector_type(4)))  _Float16 v4h;
typedef __attribute__((ext_vector_type(8)))  float    v8f;
typedef unsigned int u32x4 __attribute__((ext_vector_type(4)));
typedef int          i32x4 __attribute__((ext_vector_type(4)));
typedef int          i32x8 __attribute__((ext_vector_type(8)));

#define B_   32
#define C_   512
#define N_   1024
#define G_   32
#define CG_  16
#define EPS_ 1e-5f
#define SCALE_ 0.04419417382415922f   // 512^-0.5

#if defined(__has_builtin)
#if __has_builtin(__builtin_amdgcn_tensor_load_to_lds) && \
    __has_builtin(__builtin_amdgcn_s_wait_tensorcnt)
#define HAVE_TDM 1
#endif
#endif

// ---- workspace layout (bytes) ----
#define OFF_NT   ((size_t)0)                      // normed  f16 [B][N][C]   32 MiB
#define OFF_Q    ((size_t)32  * 1024 * 1024)      // q*scale f16 [B][N][C]   32 MiB
#define OFF_KT   ((size_t)64  * 1024 * 1024)      // k^T     f16 [B][M][C]   32 MiB
#define OFF_VT   ((size_t)96  * 1024 * 1024)      // v^T     f16 [B][C][N]   32 MiB
#define OFF_SC   ((size_t)128 * 1024 * 1024)      // scores  f16 [B][N][N]   64 MiB
#define OFF_AT   ((size_t)256 * 1024 * 1024)      // attn    f16 [B][N][N]   64 MiB
#define OFF_OUT  ((size_t)320 * 1024 * 1024)      // out     f16 [B][N][C]   32 MiB
#define OFF_WH   ((size_t)352 * 1024 * 1024)      // wq,wk,wv,wp f16          2 MiB

#define TILE_ELEMS (128 * 64)                     // halves per LDS tile buffer
#define TILE_BYTES (TILE_ELEMS * 2)

#if HAVE_TDM
// Issue one TDM 2D tile load: rows x tileW f16 elements, row stride =
// strideElems, landing contiguously ([rows][tileW] f16) at LDS byte offset
// ldsOff.  D# per CDNA5 ISA ch.8: group0 = {flags, lds_addr, global_addr,
// type=2}, group1 = {data_size=2B, tensor dims, tile dims, dim0 stride}.
__device__ __forceinline__ void tdm_load_tile(unsigned ldsOff, const void* gptr,
                                              unsigned rows, unsigned tileW,
                                              unsigned strideElems)
{
  unsigned long long ga = (unsigned long long)(size_t)gptr;
  u32x4 g0;
  g0.x = 1u;                                      // count=1, user descriptor
  g0.y = ldsOff;                                  // lds_addr (bytes)
  g0.z = (unsigned)ga;                            // global_addr[31:0]
  g0.w = (unsigned)((ga >> 32) & 0x01FFFFFFu) | (2u << 30);  // addr[56:32] | type=2
  const unsigned td0 = strideElems;               // tensor_dim0 (elements)
  const unsigned td1 = 0x00100000u;               // tensor_dim1: generous bound
  i32x8 g1;
  g1[0] = 0x00010000;                             // data_size=1 (2 bytes), mask=0
  g1[1] = (int)((td0 & 0xFFFFu) << 16);           // tensor_dim0[15:0]
  g1[2] = (int)((td0 >> 16) | ((td1 & 0xFFFFu) << 16));
  g1[3] = (int)((td1 >> 16) | (tileW << 16));     // tile_dim0 = tileW elements
  g1[4] = (int)(rows & 0xFFFFu);                  // tile_dim1 = rows, tile_dim2=0
  g1[5] = (int)strideElems;                       // tensor_dim0_stride[31:0]
  g1[6] = 0;
  g1[7] = 0;
  i32x4 zz = {0, 0, 0, 0};
#if __clang_major__ >= 23
  i32x8 z8 = {0, 0, 0, 0, 0, 0, 0, 0};
  __builtin_amdgcn_tensor_load_to_lds(g0, g1, zz, zz, z8, 0);
#else
  __builtin_amdgcn_tensor_load_to_lds(g0, g1, zz, zz, 0);
#endif
}
#endif

// Load one 16x32 f16 fragment (A-matrix VGPR layout) from a row-major
// [rows][64] LDS tile, taking K-chunk kh (0: cols 0..31, 1: cols 32..63).
// lane<16: row=lane, K={0..7,16..23}; lane>=16: row=lane-16, K={8..15,24..31}
// (per CDNA5 ISA 7.12.2).
__device__ __forceinline__ v16h ld_frag64(const _Float16* tile, int rowBase,
                                          int kh, int lane) {
  int r    = lane & 15;
  int koff = (lane >> 4) * 8;
  const _Float16* p = tile + (rowBase + r) * 64 + kh * 32 + koff;
  v8h lo = *(const v8h*)(p);
  v8h hi = *(const v8h*)(p + 16);
  return __builtin_shufflevector(lo, hi, 0,1,2,3,4,5,6,7,8,9,10,11,12,13,14,15);
}

__device__ __forceinline__ v8f wmma_f16(v16h a, v16h b, v8f c) {
  return __builtin_amdgcn_wmma_f32_16x16x32_f16(false, a, false, b,
                                                (short)0, c, false, false);
}

// C = A * B^T.  A: row-major [M][K] (lda), Bop: row-major [Ncols][K] (ldb).
// Block = 256 threads = 8 waves; block tile 128x128; wave tile 32x64
// (2 A-frags x 4 B-frags -> 8 v_wmma_f32_16x16x32_f16 per 32-wide K-chunk,
// 16 per barrier pair with the 64-deep K-step).
// TDM double-buffered: wave 0 streams A tiles, wave 1 streams B tiles,
// each gated by its own s_wait_tensorcnt.
__device__ __forceinline__ void gemm_core(
    const _Float16* __restrict__ A, int lda,
    const _Float16* __restrict__ Bop, int ldb, int K,
    int mBlk, int nBlk,
    _Float16* As,    // [2][128*64] halves
    _Float16* Bs,    // [2][128*64] halves
    v8f acc[8], int& m0, int& n0)
{
  const int tid  = threadIdx.x;
  const int lane = tid & 31;
  const int wave = tid >> 5;
  const int wm   = wave & 3;     // 4 waves down M   (4*32 = 128)
  const int wn   = wave >> 2;    // 2 waves across N (2*64 = 128)
  m0 = mBlk * 128 + wm * 32;
  n0 = nBlk * 128 + wn * 64;
#pragma unroll
  for (int i = 0; i < 8; ++i) acc[i] = 0.0f;

  const _Float16* ga = A   + (size_t)(mBlk * 128) * lda;
  const _Float16* gb = Bop + (size_t)(nBlk * 128) * ldb;

#if HAVE_TDM
  const unsigned asOff = (unsigned)(size_t)As;
  const unsigned bsOff = (unsigned)(size_t)Bs;
  const int nsteps = K / 64;
  if (wave == 0) tdm_load_tile(asOff, ga, 128, 64, lda);
  if (wave == 1) tdm_load_tile(bsOff, gb, 128, 64, ldb);
  for (int it = 0; it < nsteps; ++it) {
    const int cur = it & 1, nxt = cur ^ 1;
    const int kN  = (it + 1) * 64;
    if (wave == 0) {
      if (kN < K) {
        tdm_load_tile(asOff + (unsigned)nxt * TILE_BYTES, ga + kN, 128, 64, lda);
        __builtin_amdgcn_s_wait_tensorcnt(1);  // current A tile landed
      } else {
        __builtin_amdgcn_s_wait_tensorcnt(0);
      }
    }
    if (wave == 1) {
      if (kN < K) {
        tdm_load_tile(bsOff + (unsigned)nxt * TILE_BYTES, gb + kN, 128, 64, ldb);
        __builtin_amdgcn_s_wait_tensorcnt(1);  // current B tile landed
      } else {
        __builtin_amdgcn_s_wait_tensorcnt(0);
      }
    }
    __syncthreads();                           // release compute waves
    const _Float16* Ab = As + cur * TILE_ELEMS;
    const _Float16* Bb = Bs + cur * TILE_ELEMS;
#pragma unroll
    for (int kh = 0; kh < 2; ++kh) {
      v16h a0 = ld_frag64(Ab, wm * 32,      kh, lane);
      v16h a1 = ld_frag64(Ab, wm * 32 + 16, kh, lane);
      v16h b0 = ld_frag64(Bb, wn * 64,      kh, lane);
      v16h b1 = ld_frag64(Bb, wn * 64 + 16, kh, lane);
      v16h b2 = ld_frag64(Bb, wn * 64 + 32, kh, lane);
      v16h b3 = ld_frag64(Bb, wn * 64 + 48, kh, lane);
      acc[0] = wmma_f16(a0, b0, acc[0]);
      acc[1] = wmma_f16(a0, b1, acc[1]);
      acc[2] = wmma_f16(a0, b2, acc[2]);
      acc[3] = wmma_f16(a0, b3, acc[3]);
      acc[4] = wmma_f16(a1, b0, acc[4]);
      acc[5] = wmma_f16(a1, b1, acc[5]);
      acc[6] = wmma_f16(a1, b2, acc[6]);
      acc[7] = wmma_f16(a1, b3, acc[7]);
    }
    __syncthreads();                           // buffer fully consumed
  }
#else
  // Fallback: register staging through LDS (single buffer, two barriers).
  for (int k0 = 0; k0 < K; k0 += 64) {
    v8h ar[4], br[4];
#pragma unroll
    for (int j = 0; j < 4; ++j) {
      int chunk = tid + j * 256;               // 1024 chunks of 8 halves
      int row = chunk >> 3, off = (chunk & 7) * 8;
      ar[j] = *(const v8h*)(ga + (size_t)row * lda + k0 + off);
      br[j] = *(const v8h*)(gb + (size_t)row * ldb + k0 + off);
    }
    __syncthreads();
#pragma unroll
    for (int j = 0; j < 4; ++j) {
      int chunk = tid + j * 256;
      int row = chunk >> 3, off = (chunk & 7) * 8;
      *(v8h*)(As + row * 64 + off) = ar[j];
      *(v8h*)(Bs + row * 64 + off) = br[j];
    }
    __syncthreads();
#pragma unroll
    for (int kh = 0; kh < 2; ++kh) {
      v16h a0 = ld_frag64(As, wm * 32,      kh, lane);
      v16h a1 = ld_frag64(As, wm * 32 + 16, kh, lane);
      v16h b0 = ld_frag64(Bs, wn * 64,      kh, lane);
      v16h b1 = ld_frag64(Bs, wn * 64 + 16, kh, lane);
      v16h b2 = ld_frag64(Bs, wn * 64 + 32, kh, lane);
      v16h b3 = ld_frag64(Bs, wn * 64 + 48, kh, lane);
      acc[0] = wmma_f16(a0, b0, acc[0]);
      acc[1] = wmma_f16(a0, b1, acc[1]);
      acc[2] = wmma_f16(a0, b2, acc[2]);
      acc[3] = wmma_f16(a0, b3, acc[3]);
      acc[4] = wmma_f16(a1, b0, acc[4]);
      acc[5] = wmma_f16(a1, b1, acc[5]);
      acc[6] = wmma_f16(a1, b2, acc[6]);
      acc[7] = wmma_f16(a1, b3, acc[7]);
    }
  }
#endif
}

// ---- weights f32 -> f16 (wq|wk|wv|wp concatenated) ----
__global__ __launch_bounds__(256)
void k_w2h(const float* __restrict__ wq, const float* __restrict__ wk,
           const float* __restrict__ wv, const float* __restrict__ wp,
           _Float16* __restrict__ wh)
{
  int i = blockIdx.x * 256 + threadIdx.x;
  if (i >= 4 * C_ * C_) return;
  int w = i / (C_ * C_), r = i % (C_ * C_);
  const float* src = (w == 0) ? wq : (w == 1) ? wk : (w == 2) ? wv : wp;
  wh[i] = (_Float16)src[r];
}

// ---- GroupNorm: one block per (b, g); writes normed^T f16 [B][N][C].
// Normalized values are repacked through a [n][16] LDS tile so the
// transposed global writes go out as two b128 stores per n instead of
// sixteen scattered b16 stores. ----
__global__ __launch_bounds__(256)
void k_gn(const float* __restrict__ x, const float* __restrict__ gs,
          const float* __restrict__ gb, _Float16* __restrict__ nT)
{
  __shared__ float rs[8], rq[8], stat[2];
  __shared__ _Float16 lt[CG_ * N_];            // [n][cl], 32 KiB
  const int bg = blockIdx.x, b = bg >> 5, g = bg & 31;
  const float* xp = x + ((size_t)b * C_ + g * CG_) * N_;
  const int tid = threadIdx.x;
  float s = 0.f, q = 0.f;
  for (int i = tid; i < CG_ * N_; i += 256) { float t = xp[i]; s += t; q += t * t; }
  for (int off = 16; off; off >>= 1) {
    s += __shfl_xor(s, off, 32);
    q += __shfl_xor(q, off, 32);
  }
  if ((tid & 31) == 0) { rs[tid >> 5] = s; rq[tid >> 5] = q; }
  __syncthreads();
  if (tid == 0) {
    float S = 0.f, Q = 0.f;
    for (int i = 0; i < 8; ++i) { S += rs[i]; Q += rq[i]; }
    float mean = S / (float)(CG_ * N_);
    float var  = Q / (float)(CG_ * N_) - mean * mean;
    stat[0] = mean; stat[1] = rsqrtf(var + EPS_);
  }
  __syncthreads();
  const float mean = stat[0], inv = stat[1];
  for (int i = tid; i < CG_ * N_; i += 256) {
    int cl = i >> 10, n = i & (N_ - 1);
    int c  = g * CG_ + cl;
    float t = (xp[i] - mean) * inv * gs[c] + gb[c];
    lt[n * CG_ + cl] = (_Float16)t;
  }
  __syncthreads();
#pragma unroll
  for (int j = 0; j < 4; ++j) {
    int n = tid + j * 256;
    v8h lo = *(const v8h*)(lt + n * CG_);
    v8h hi = *(const v8h*)(lt + n * CG_ + 8);
    _Float16* dst = nT + ((size_t)b * N_ + n) * C_ + g * CG_;
    *(v8h*)(dst)     = lo;
    *(v8h*)(dst + 8) = hi;
  }
}

// ---- Q / K^T / V GEMMs: P = nT * W^T + bias (q pre-scaled by C^-0.5) ----
__global__ __launch_bounds__(256)
void k_qkv(const _Float16* __restrict__ nT, const _Float16* __restrict__ wh,
           const float* __restrict__ bq, const float* __restrict__ bk,
           const float* __restrict__ bv,
           _Float16* __restrict__ q, _Float16* __restrict__ kT,
           _Float16* __restrict__ vT)
{
  __shared__ _Float16 As[2 * TILE_ELEMS], Bs[2 * TILE_ELEMS];
  const int which = blockIdx.z % 3;
  const int b     = blockIdx.z / 3;
  const _Float16* A = nT + (size_t)b * N_ * C_;
  const _Float16* W = wh + (size_t)which * C_ * C_;
  const float* bias = (which == 0) ? bq : (which == 1) ? bk : bv;
  const float postScale = (which == 0) ? SCALE_ : 1.0f;
  v8f acc[8]; int m0, n0;
  gemm_core(A, C_, W, C_, C_, blockIdx.x, blockIdx.y, As, Bs, acc, m0, n0);

  const int lane = threadIdx.x & 31;
  const int colb = lane & 15, rowAdd = (lane >> 4) * 8;
  _Float16* qk = (which == 0) ? q : kT;
  for (int r = 0; r < 8; ++r) {
#pragma unroll
    for (int xx = 0; xx < 2; ++xx) {
      int row = m0 + r + rowAdd + xx * 16;
#pragma unroll
      for (int yy = 0; yy < 4; ++yy) {
        int c = n0 + colb + yy * 16;
        float v = (acc[xx * 4 + yy][r] + bias[c]) * postScale;
        if (which != 2) {
          qk[((size_t)b * N_ + row) * C_ + c] = (_Float16)v;
        } else {  // v stored transposed [B][C][N] so attn*V is K-contiguous
          vT[((size_t)b * C_ + c) * N_ + row] = (_Float16)v;
        }
      }
    }
  }
}

// ---- scores = (q*scale) * kT^T -> f16 (pre-softmax logits) ----
__global__ __launch_bounds__(256)
void k_scores(const _Float16* __restrict__ q, const _Float16* __restrict__ kT,
              _Float16* __restrict__ sc)
{
  __shared__ _Float16 As[2 * TILE_ELEMS], Bs[2 * TILE_ELEMS];
  const int b = blockIdx.z;
  v8f acc[8]; int m0, n0;
  gemm_core(q  + (size_t)b * N_ * C_, C_,
            kT + (size_t)b * N_ * C_, C_, C_,
            blockIdx.x, blockIdx.y, As, Bs, acc, m0, n0);
  const int lane = threadIdx.x & 31;
  const int colb = lane & 15, rowAdd = (lane >> 4) * 8;
  _Float16* base = sc + (size_t)b * N_ * N_;
  for (int r = 0; r < 8; ++r) {
#pragma unroll
    for (int xx = 0; xx < 2; ++xx) {
      int row = m0 + r + rowAdd + xx * 16;
#pragma unroll
      for (int yy = 0; yy < 4; ++yy) {
        int c = n0 + colb + yy * 16;
        base[(size_t)row * N_ + c] = (_Float16)acc[xx * 4 + yy][r];
      }
    }
  }
}

// ---- softmax over rows of 1024, f16 -> f16 ----
__global__ __launch_bounds__(256)
void k_softmax(const _Float16* __restrict__ sc, _Float16* __restrict__ attn)
{
  __shared__ float red[8];
  const size_t row = blockIdx.x;             // b*1024 + n
  const _Float16* p = sc + row * N_;
  const int tid = threadIdx.x;
  v4h d = *(const v4h*)(p + tid * 4);        // packed b64 load
  float v[4];
  float m = -3.4e38f;
#pragma unroll
  for (int j = 0; j < 4; ++j) { v[j] = (float)d[j]; m = fmaxf(m, v[j]); }
  for (int off = 16; off; off >>= 1) m = fmaxf(m, __shfl_xor(m, off, 32));
  if ((tid & 31) == 0) red[tid >> 5] = m;
  __syncthreads();
  float bm = red[0];
  for (int i = 1; i < 8; ++i) bm = fmaxf(bm, red[i]);
  float s = 0.f;
#pragma unroll
  for (int j = 0; j < 4; ++j) { v[j] = __expf(v[j] - bm); s += v[j]; }
  for (int off = 16; off; off >>= 1) s += __shfl_xor(s, off, 32);
  __syncthreads();
  if ((tid & 31) == 0) red[tid >> 5] = s;
  __syncthreads();
  float bs = 0.f;
  for (int i = 0; i < 8; ++i) bs += red[i];
  const float inv = 1.f / bs;
  v4h o;
#pragma unroll
  for (int j = 0; j < 4; ++j) o[j] = (_Float16)(v[j] * inv);
  *(v4h*)(attn + row * N_ + tid * 4) = o;    // packed b64 store
}

// ---- out = attn * v  (via vT) -> f16 [B][N][C] ----
__global__ __launch_bounds__(256)
void k_out(const _Float16* __restrict__ attn, const _Float16* __restrict__ vT,
           _Float16* __restrict__ outb)
{
  __shared__ _Float16 As[2 * TILE_ELEMS], Bs[2 * TILE_ELEMS];
  const int b = blockIdx.z;
  v8f acc[8]; int m0, n0;
  gemm_core(attn + (size_t)b * N_ * N_, N_,
            vT   + (size_t)b * C_ * N_, N_, N_,
            blockIdx.x, blockIdx.y, As, Bs, acc, m0, n0);
  const int lane = threadIdx.x & 31;
  const int colb = lane & 15, rowAdd = (lane >> 4) * 8;
  _Float16* base = outb + (size_t)b * N_ * C_;
  for (int r = 0; r < 8; ++r) {
#pragma unroll
    for (int xx = 0; xx < 2; ++xx) {
      int row = m0 + r + rowAdd + xx * 16;
#pragma unroll
      for (int yy = 0; yy < 4; ++yy) {
        int c = n0 + colb + yy * 16;
        base[(size_t)row * C_ + c] = (_Float16)acc[xx * 4 + yy][r];
      }
    }
  }
}

// ---- y = x + wp * out^T + bp ----
__global__ __launch_bounds__(256)
void k_proj(const _Float16* __restrict__ wph, const _Float16* __restrict__ outb,
            const float* __restrict__ bp, const float* __restrict__ x,
            float* __restrict__ y)
{
  __shared__ _Float16 As[2 * TILE_ELEMS], Bs[2 * TILE_ELEMS];
  const int b = blockIdx.z;
  v8f acc[8]; int m0, n0;
  gemm_core(wph, C_, outb + (size_t)b * N_ * C_, C_, C_,
            blockIdx.x, blockIdx.y, As, Bs, acc, m0, n0);
  const int lane = threadIdx.x & 31;
  const int colb = lane & 15, rowAdd = (lane >> 4) * 8;
  for (int r = 0; r < 8; ++r) {
#pragma unroll
    for (int xx = 0; xx < 2; ++xx) {
      int o = m0 + r + rowAdd + xx * 16;       // output channel
      float bb = bp[o];
#pragma unroll
      for (int yy = 0; yy < 4; ++yy) {
        size_t i = ((size_t)b * C_ + o) * N_ + n0 + colb + yy * 16;
        y[i] = x[i] + acc[xx * 4 + yy][r] + bb;
      }
    }
  }
}

extern "C" void kernel_launch(void* const* d_in, const int* in_sizes, int n_in,
                              void* d_out, int out_size, void* d_ws, size_t ws_size,
                              hipStream_t stream) {
  (void)in_sizes; (void)n_in; (void)out_size; (void)ws_size;
  const float* x  = (const float*)d_in[0];
  const float* gs = (const float*)d_in[1];
  const float* gbi= (const float*)d_in[2];
  const float* wq = (const float*)d_in[3];
  const float* bq = (const float*)d_in[4];
  const float* wk = (const float*)d_in[5];
  const float* bk = (const float*)d_in[6];
  const float* wv = (const float*)d_in[7];
  const float* bv = (const float*)d_in[8];
  const float* wp = (const float*)d_in[9];
  const float* bp = (const float*)d_in[10];
  float* y = (float*)d_out;

  char* ws = (char*)d_ws;
  _Float16* nT   = (_Float16*)(ws + OFF_NT);
  _Float16* q    = (_Float16*)(ws + OFF_Q);
  _Float16* kT   = (_Float16*)(ws + OFF_KT);
  _Float16* vT   = (_Float16*)(ws + OFF_VT);
  _Float16* sc   = (_Float16*)(ws + OFF_SC);
  _Float16* attn = (_Float16*)(ws + OFF_AT);
  _Float16* outb = (_Float16*)(ws + OFF_OUT);
  _Float16* wh   = (_Float16*)(ws + OFF_WH);
  _Float16* wph  = wh + (size_t)3 * C_ * C_;

  k_w2h    <<<dim3(4 * C_ * C_ / 256), 256, 0, stream>>>(wq, wk, wv, wp, wh);
  k_gn     <<<dim3(B_ * G_),           256, 0, stream>>>(x, gs, gbi, nT);
  k_qkv    <<<dim3(N_ / 128, C_ / 128, B_ * 3), 256, 0, stream>>>(nT, wh, bq, bk, bv, q, kT, vT);
  k_scores <<<dim3(N_ / 128, N_ / 128, B_),     256, 0, stream>>>(q, kT, sc);
  k_softmax<<<dim3(B_ * N_),                    256, 0, stream>>>(sc, attn);
  k_out    <<<dim3(N_ / 128, C_ / 128, B_),     256, 0, stream>>>(attn, vT, outb);
  k_proj   <<<dim3(C_ / 128, N_ / 128, B_),     256, 0, stream>>>(wph, outb, bp, x, y);
}
